// Nl_88192858456666
// MI455X (gfx1250) — matrix operand order repaired and checked by reading, not compiled, not run on hardware
//
#include <hip/hip_runtime.h>
#include <hip/hip_fp16.h>
#include <stdint.h>

// ---------------------------------------------------------------------------
// Attention block (B=4, C=256, H=W=64 -> N=4096, KC=128, VC=OC=256)
// fp16 WMMA (v_wmma_f32_16x16x32_f16) everywhere, fp32 accumulation,
// flash-attention online softmax; K/V tiles double-buffered in LDS via
// CDNA5 async global->LDS loads (ASYNCcnt).
// ---------------------------------------------------------------------------

typedef __attribute__((ext_vector_type(16))) _Float16 v16h;
typedef __attribute__((ext_vector_type(8)))  float    v8f;
typedef int v4i_gnu __attribute__((vector_size(16)));   // matches builtin param

#define WMMA16(a, b, c) \
  __builtin_amdgcn_wmma_f32_16x16x32_f16(false, (a), false, (b), (short)0, (c), false, false)

union F16x16 { uint4 q[2]; v16h h; };
union H8 { _Float16 h[8]; uint4 q; };

#define NB   4
#define NC   256
#define NN   4096
#define NKC  128
#define NVC  256

// ---- CDNA5 async global->LDS staging (guarded; sync-copy fallback) --------
#if __has_builtin(__builtin_amdgcn_global_load_async_to_lds_b128)
#define HAVE_ASYNC_LDS 1
// AS1 via 64-bit int round-trip; AS3 via 32-bit truncation of the flat addr
// (LDS aperture: flat addr[31:0] == LDS byte address).
#define STAGE_B128(gsrc, ldst)                                                 \
  __builtin_amdgcn_global_load_async_to_lds_b128(                              \
      (__attribute__((address_space(1))) v4i_gnu*)(uintptr_t)(gsrc),           \
      (__attribute__((address_space(3))) v4i_gnu*)(uint32_t)(uintptr_t)(ldst), \
      0, 0)
#if __has_builtin(__builtin_amdgcn_s_wait_asynccnt)
#define WAIT_ASYNC(n) __builtin_amdgcn_s_wait_asynccnt(n)
#else
#define WAIT_ASYNC(n) asm volatile("s_wait_asynccnt %0" ::"n"(n) : "memory")
#endif
#else
#define STAGE_B128(gsrc, ldst) *(uint4*)(ldst) = *(const uint4*)(gsrc)
#define WAIT_ASYNC(n)
#endif

// ---------------------------------------------------------------------------
// Kernel 0: convert weights fp32 -> fp16.  wkvh = [wk(128 rows); wv(256 rows)]
// ---------------------------------------------------------------------------
__global__ void k_convert_w(const float* __restrict__ wk, const float* __restrict__ wv,
                            const float* __restrict__ ww,
                            _Float16* __restrict__ wkvh, _Float16* __restrict__ wwh) {
  int i = blockIdx.x * 256 + threadIdx.x;           // 0 .. 98303
  if (i < 128 * 256)       wkvh[i] = (_Float16)wk[i];
  else if (i < 384 * 256)  wkvh[i] = (_Float16)wv[i - 128 * 256];
  if (i < 256 * 256)       wwh[i]  = (_Float16)ww[i];
}

// ---------------------------------------------------------------------------
// Kernel 1: fused K/V projection.
//   Kmat[b][n][k] = sum_c wk[k][c]*x[b][c][n] + bk[k]        ([N][128] f16)
//   Vt  [b][v][n] = sum_c wv[v][c]*x[b][c][n] + bv[v]        ([256][N] f16)
// Block = 128 thr (4 waves), 64-token tile; wave w owns tokens [16w,16w+16).
// xT tile (32c x 64n) staged transposed in LDS (f32->f16 through VGPRs).
// ---------------------------------------------------------------------------
__global__ __launch_bounds__(128)
void k_proj(const float* __restrict__ x, const _Float16* __restrict__ wkvh,
            const float* __restrict__ bk, const float* __restrict__ bv,
            _Float16* __restrict__ Kmat, _Float16* __restrict__ Vt) {
  const int b  = blockIdx.x >> 6;
  const int n0 = (blockIdx.x & 63) * 64;
  const int t  = threadIdx.x;
  const int wid = t >> 5, lane = t & 31;
  const int lm = lane & 15, lh = lane >> 4;

  __shared__ _Float16 xT[64][40];                   // [token][c], 80B rows (16B mult)

  v8f acc[24];
  v8f vzero = {};
  for (int o = 0; o < 24; ++o) acc[o] = vzero;

  const float* xb = x + (size_t)b * NC * NN;

  for (int c0 = 0; c0 < NC; c0 += 32) {
    __syncthreads();
    for (int it = 0; it < 16; ++it) {               // stage 32c x 64n, transposed
      int idx = it * 128 + t;                       // 0..2047
      int cc = idx >> 6, nn = idx & 63;
      xT[nn][cc] = (_Float16)xb[(size_t)(c0 + cc) * NN + n0 + nn];
    }
    __syncthreads();

    // A fragment: rows = this wave's 16 tokens, k = 32 channels of this step
    F16x16 A;
    const _Float16* ar = &xT[16 * wid + lm][0];
    A.q[0] = *(const uint4*)(ar + 16 * lh);
    A.q[1] = *(const uint4*)(ar + 16 * lh + 8);

    for (int o = 0; o < 24; ++o) {                  // 384 output channels
      F16x16 Bf;
      const _Float16* br = wkvh + (size_t)(16 * o + lm) * NC + c0 + 16 * lh;
      Bf.q[0] = *(const uint4*)br;
      Bf.q[1] = *(const uint4*)(br + 8);
      acc[o] = WMMA16(A.h, Bf.h, acc[o]);
    }
  }

  // Epilogue: D element (m, ncol) -> lane = ncol + 16*(m/8), vgpr r = m%8
  for (int o = 0; o < 24; ++o) {
    if (o < 8) {                                    // key channels -> Kmat [n][128]
      int k = 16 * o + lm;
      float bias = bk[k];
      for (int r = 0; r < 8; ++r) {
        int tok = n0 + 16 * wid + r + 8 * lh;
        Kmat[((size_t)b * NN + tok) * NKC + k] = (_Float16)(acc[o][r] + bias);
      }
    } else {                                        // value channels -> Vt [v][n]
      int v = 16 * (o - 8) + lm;
      float bias = bv[v];
      H8 tmp;
      for (int r = 0; r < 8; ++r) tmp.h[r] = (_Float16)(acc[o][r] + bias);
      _Float16* dst = Vt + ((size_t)b * NVC + v) * NN + n0 + 16 * wid + 8 * lh;
      *(uint4*)dst = tmp.q;                         // 8 consecutive n, 16B aligned
    }
  }
}

// ---------------------------------------------------------------------------
// Kernel 2: flash attention.  64 query rows / block (4 waves x 16 rows),
// 64-key steps, double-buffered async K/V tile staging.  ctx[b][n][v] f16.
// ---------------------------------------------------------------------------
__global__ __launch_bounds__(128)
void k_attn(const _Float16* __restrict__ Kmat, const _Float16* __restrict__ Vt,
            _Float16* __restrict__ ctx) {
  const int b  = blockIdx.x >> 6;
  const int n0 = (blockIdx.x & 63) * 64;
  const int t  = threadIdx.x;
  const int wid = t >> 5, lane = t & 31;
  const int lm = lane & 15, lh = lane >> 4;

  __shared__ _Float16 KT[2][64][136];               // keys   [j][k]   272B rows
  __shared__ _Float16 VT[2][256][72];               // values [v][j]   144B rows
  __shared__ _Float16 PB[4][16][72];                // per-wave P bounce buffer

  const _Float16* Kb = Kmat + (size_t)b * NN * NKC;
  const _Float16* Vb = Vt   + (size_t)b * NVC * NN;

  // issue one step's K/V tile staging (24 x b128 per thread)
  auto stage = [&](int buf, int j0) {
    for (int it = 0; it < 8; ++it) {                // K tile: 64 x 128 halves
      int idx = it * 128 + t;                       // b128 chunk index
      int row = idx >> 4, col = (idx & 15) * 8;
      STAGE_B128(Kb + (size_t)(j0 + row) * NKC + col, &KT[buf][row][col]);
    }
    for (int it = 0; it < 16; ++it) {               // V tile: 256 x 64 halves
      int idx = it * 128 + t;
      int row = idx >> 3, col = (idx & 7) * 8;
      STAGE_B128(Vb + (size_t)row * NN + j0 + col, &VT[buf][row][col]);
    }
  };

  stage(0, 0);                                      // prologue: step 0 in flight

  // Q fragments: 16 rows x 128 k, resident for the whole kernel
  F16x16 Qf[4];
  {
    const _Float16* qr = Kb + (size_t)(n0 + 16 * wid + lm) * NKC;
    for (int kk = 0; kk < 4; ++kk) {
      Qf[kk].q[0] = *(const uint4*)(qr + 32 * kk + 16 * lh);
      Qf[kk].q[1] = *(const uint4*)(qr + 32 * kk + 16 * lh + 8);
    }
  }

  v8f O[16];
  v8f vzero = {};
  for (int o = 0; o < 16; ++o) O[o] = vzero;
  float mrow[8], lrow[8];
  for (int r = 0; r < 8; ++r) { mrow[r] = -__builtin_inff(); lrow[r] = 0.f; }

  const int NSTEP = NN / 64;
  for (int s = 0; s < NSTEP; ++s) {
    const int buf = s & 1;
    if (s + 1 < NSTEP) {                            // prefetch next tile
      stage(buf ^ 1, (s + 1) * 64);
      WAIT_ASYNC(24);                               // step-s loads complete (in order)
    } else {
      WAIT_ASYNC(0);
    }
    __syncthreads();                                // tile for step s visible to all

    // S = Q * K^T   (4 subtiles of 16 key columns)
    v8f S[4];
    for (int sc = 0; sc < 4; ++sc) {
      v8f c = vzero;
      for (int kk = 0; kk < 4; ++kk) {
        F16x16 Bf;
        const _Float16* kr = &KT[buf][16 * sc + lm][32 * kk + 16 * lh];
        Bf.q[0] = *(const uint4*)kr;
        Bf.q[1] = *(const uint4*)(kr + 8);
        c = WMMA16(Qf[kk].h, Bf.h, c);
      }
      S[sc] = c;
    }

    // Online softmax: row r lives on the 16 lanes of this half-wave
    float alpha[8];
    for (int r = 0; r < 8; ++r) {
      float rm = fmaxf(fmaxf(S[0][r], S[1][r]), fmaxf(S[2][r], S[3][r]));
      for (int off = 1; off < 16; off <<= 1) rm = fmaxf(rm, __shfl_xor(rm, off, 32));
      float mnew = fmaxf(mrow[r], rm);
      float a = __expf(mrow[r] - mnew);
      float ls = 0.f;
      for (int sc = 0; sc < 4; ++sc) {
        float p = __expf(S[sc][r] - mnew);
        S[sc][r] = p;
        ls += p;
      }
      for (int off = 1; off < 16; off <<= 1) ls += __shfl_xor(ls, off, 32);
      lrow[r] = lrow[r] * a + ls;
      mrow[r] = mnew;
      alpha[r] = a;
    }
    for (int o = 0; o < 16; ++o)
      for (int r = 0; r < 8; ++r) O[o][r] *= alpha[r];

    // Bounce P through LDS: C/D layout -> A-fragment layout
    for (int sc = 0; sc < 4; ++sc)
      for (int r = 0; r < 8; ++r)
        PB[wid][r + 8 * lh][16 * sc + lm] = (_Float16)S[sc][r];
    __syncthreads();

    F16x16 Pf[2];
    {
      const _Float16* pr = &PB[wid][lm][0];
      for (int kk = 0; kk < 2; ++kk) {
        Pf[kk].q[0] = *(const uint4*)(pr + 32 * kk + 16 * lh);
        Pf[kk].q[1] = *(const uint4*)(pr + 32 * kk + 16 * lh + 8);
      }
    }

    // O += P * V   (16 v-subtiles x 2 k-steps)
    for (int o = 0; o < 16; ++o) {
      F16x16 Bf;
      const _Float16* vr = &VT[buf][16 * o + lm][16 * lh];
      Bf.q[0] = *(const uint4*)vr;
      Bf.q[1] = *(const uint4*)(vr + 8);
      v8f c = WMMA16(Pf[0].h, Bf.h, O[o]);
      const _Float16* vr2 = &VT[buf][16 * o + lm][32 + 16 * lh];
      Bf.q[0] = *(const uint4*)vr2;
      Bf.q[1] = *(const uint4*)(vr2 + 8);
      O[o] = WMMA16(Pf[1].h, Bf.h, c);
    }
    __syncthreads();                                // buf fully consumed; safe to refill
  }

  float inv[8];
  for (int r = 0; r < 8; ++r) inv[r] = 1.f / lrow[r];
  for (int o = 0; o < 16; ++o)
    for (int r = 0; r < 8; ++r) {
      int tok = n0 + 16 * wid + r + 8 * lh;
      ctx[((size_t)b * NN + tok) * NVC + 16 * o + lm] = (_Float16)(O[o][r] * inv[r]);
    }
}

// ---------------------------------------------------------------------------
// Kernel 3: out[b][o][n] = ww @ ctx^T + wb + x   (fp32 output, residual add)
// ---------------------------------------------------------------------------
__global__ __launch_bounds__(128)
void k_out(const _Float16* __restrict__ ctx, const _Float16* __restrict__ wwh,
           const float* __restrict__ wb, const float* __restrict__ x,
           float* __restrict__ out) {
  const int b  = blockIdx.x >> 6;
  const int n0 = (blockIdx.x & 63) * 64;
  const int t  = threadIdx.x;
  const int wid = t >> 5, lane = t & 31;
  const int lm = lane & 15, lh = lane >> 4;

  v8f acc[16];
  v8f vzero = {};
  for (int o = 0; o < 16; ++o) acc[o] = vzero;

  const _Float16* cr = ctx + ((size_t)b * NN + n0 + 16 * wid + lm) * NVC;
  for (int kk = 0; kk < 8; ++kk) {                  // 256 v channels, 32 / step
    F16x16 A;
    A.q[0] = *(const uint4*)(cr + 32 * kk + 16 * lh);
    A.q[1] = *(const uint4*)(cr + 32 * kk + 16 * lh + 8);
    for (int o = 0; o < 16; ++o) {
      F16x16 Bf;
      const _Float16* wr = wwh + (size_t)(16 * o + lm) * NVC + 32 * kk + 16 * lh;
      Bf.q[0] = *(const uint4*)wr;
      Bf.q[1] = *(const uint4*)(wr + 8);
      acc[o] = WMMA16(A.h, Bf.h, acc[o]);
    }
  }

  for (int o = 0; o < 16; ++o) {
    int och = 16 * o + lm;
    float bias = wb[och];
    const float* xr   = x   + ((size_t)b * 256 + och) * NN + n0 + 16 * wid + 8 * lh;
    float*       orow = out + ((size_t)b * 256 + och) * NN + n0 + 16 * wid + 8 * lh;
    float4 x0 = *(const float4*)xr;
    float4 x1 = *(const float4*)(xr + 4);
    float4 r0, r1;
    r0.x = acc[o][0] + bias + x0.x;  r0.y = acc[o][1] + bias + x0.y;
    r0.z = acc[o][2] + bias + x0.z;  r0.w = acc[o][3] + bias + x0.w;
    r1.x = acc[o][4] + bias + x1.x;  r1.y = acc[o][5] + bias + x1.y;
    r1.z = acc[o][6] + bias + x1.z;  r1.w = acc[o][7] + bias + x1.w;
    *(float4*)orow = r0;
    *(float4*)(orow + 4) = r1;
  }
}

// ---------------------------------------------------------------------------
extern "C" void kernel_launch(void* const* d_in, const int* in_sizes, int n_in,
                              void* d_out, int out_size, void* d_ws, size_t ws_size,
                              hipStream_t stream) {
  const float* x  = (const float*)d_in[0];
  const float* wk = (const float*)d_in[1];
  const float* bk = (const float*)d_in[2];
  const float* wv = (const float*)d_in[3];
  const float* bv = (const float*)d_in[4];
  const float* ww = (const float*)d_in[5];
  const float* wb = (const float*)d_in[6];
  float* out = (float*)d_out;

  // workspace layout (halves), total ~21.3 MB
  _Float16* wkvh = (_Float16*)d_ws;                 // 384*256
  _Float16* wwh  = wkvh + 384 * 256;                // 256*256
  _Float16* Kmat = wwh  + 256 * 256;                // B*N*128
  _Float16* Vt   = Kmat + (size_t)NB * NN * NKC;    // B*256*N
  _Float16* ctx  = Vt   + (size_t)NB * NVC * NN;    // B*N*256

  k_convert_w<<<384, 256, 0, stream>>>(wk, wv, ww, wkvh, wwh);
  k_proj<<<NB * 64, 128, 0, stream>>>(x, wkvh, bk, bv, Kmat, Vt);
  k_attn<<<NB * 64, 128, 0, stream>>>(Kmat, Vt, ctx);
  k_out <<<NB * 64, 128, 0, stream>>>(ctx, wwh, wb, x, out);
}
